// QNetwork_73538430042520
// MI455X (gfx1250) — matrix operand order repaired and checked by reading, not compiled, not run on hardware
//
#include <hip/hip_runtime.h>
#include <hip/hip_bf16.h>

// MI455X / gfx1250 implementation.
// Roofline: edge scatter moves ~3.2 GB (>=137us @ 23.3 TB/s) and dominates;
// all GEMMs (~13 GFLOP) are routed through v_wmma_f32_16x16x32_bf16 so they
// are effectively free. Weights are converted to bf16 once (128 KB, L2-resident).

typedef __attribute__((ext_vector_type(16))) __bf16 v16bf;
typedef __attribute__((ext_vector_type(8)))  __bf16 v8bf;
typedef __attribute__((ext_vector_type(4)))  __bf16 v4bf;
typedef __attribute__((ext_vector_type(8)))  float  v8f;
typedef __attribute__((ext_vector_type(4)))  float  v4f;

#define NP 128   // P
#define NK 128   // K
#define N2P 256  // 2P

// ---------------------------------------------------------------- helpers
__device__ __forceinline__ float relu_f(float x) { return x > 0.f ? x : 0.f; }

__device__ __forceinline__ void put4(v16bf& a, int o, v4f v) {
  a[o + 0] = (__bf16)v.x; a[o + 1] = (__bf16)v.y;
  a[o + 2] = (__bf16)v.z; a[o + 3] = (__bf16)v.w;
}
__device__ __forceinline__ v4f relu4(v4f v) {
  v.x = relu_f(v.x); v.y = relu_f(v.y); v.z = relu_f(v.z); v.w = relu_f(v.w);
  return v;
}

// ---------------------------------------------------------------- kernels
// f32 -> bf16 weight conversion (theta2, theta5, h_theta)
__global__ void k_cvt_bf16(const float* __restrict__ src, __bf16* __restrict__ dst, int n) {
  int i = blockIdx.x * blockDim.x + threadIdx.x;
  if (i < n) dst[i] = (__bf16)src[i];
}

// Row-normalize x -> bf16 xn. One wave per row, 4 f32 per lane.
__global__ void k_norm(const float* __restrict__ x, __bf16* __restrict__ xn, int n) {
  int wave = (blockIdx.x * blockDim.x + threadIdx.x) >> 5;
  int lane = threadIdx.x & 31;
  if (wave >= n) return;
  v4f v = *(const v4f*)(x + (size_t)wave * NK + lane * 4);
  float ss = v.x * v.x + v.y * v.y + v.z * v.z + v.w * v.w;
  #pragma unroll
  for (int m = 16; m >= 1; m >>= 1) ss += __shfl_xor(ss, m, 32);
  float nrm = sqrtf(ss);
  float inv = (nrm == 0.f) ? 1.f : 1.f / nrm;
  v4bf o;
  o.x = (__bf16)(v.x * inv); o.y = (__bf16)(v.y * inv);
  o.z = (__bf16)(v.z * inv); o.w = (__bf16)(v.w * inv);
  *(v4bf*)(xn + (size_t)wave * NK + lane * 4) = o;
}

// mu = relu(conn @ theta1^T + xn @ theta2^T). One wave per 16-node tile.
// A-tile (16x32 bf16): lane<16 -> row=lane, K {0..15}; lane>=16 -> K {16..31}
// B-tile: lane holds weight row (=output column), 16 contiguous bf16 along K.
__global__ void k_mu(const float* __restrict__ s_v, const float* __restrict__ t_v,
                     const __bf16* __restrict__ xn, const __bf16* __restrict__ t2,
                     const float* __restrict__ th1, float* __restrict__ mu, int ntiles) {
  int tile = blockIdx.x * (blockDim.x >> 5) + (threadIdx.x >> 5);
  if (tile >= ntiles) return;
  int lane = threadIdx.x & 31;
  int half = lane >> 4, l15 = lane & 15;
  int m0 = tile * 16;
  int rowA = m0 + l15;
  v8f acc[8] = {};
  #pragma unroll
  for (int kc = 0; kc < 4; ++kc) {
    int k0 = kc * 32;
    const v8bf* pa = (const v8bf*)(xn + (size_t)rowA * NK + k0 + half * 16);
    v16bf a = __builtin_shufflevector(pa[0], pa[1],
        0,1,2,3,4,5,6,7,8,9,10,11,12,13,14,15);
    #pragma unroll
    for (int t = 0; t < 8; ++t) {
      v16bf b = *(const v16bf*)(t2 + (size_t)(t * 16 + l15) * NK + k0 + half * 16);
      acc[t] = __builtin_amdgcn_wmma_f32_16x16x32_bf16(
          false, a, false, b, (short)0, acc[t], false, false);
    }
  }
  // epilogue: add conn @ theta1^T, relu, store f32 mu
  int rbase = m0 + half * 8;
  float sv[8], tv[8];
  #pragma unroll
  for (int r = 0; r < 8; ++r) { sv[r] = s_v[rbase + r]; tv[r] = t_v[rbase + r]; }
  #pragma unroll
  for (int t = 0; t < 8; ++t) {
    int c = t * 16 + l15;
    float a1 = th1[c * 2 + 0], b1 = th1[c * 2 + 1];
    #pragma unroll
    for (int r = 0; r < 8; ++r) {
      float val = relu_f(acc[t][r] + sv[r] * a1 + tv[r] * b1);
      mu[(size_t)(rbase + r) * NP + c] = val;
    }
  }
}

// Edge scatter: nb[src,:] += mu[dst,:]; deg[src] += 1. One wave per edge.
__global__ void k_scatter(const int* __restrict__ esrc, const int* __restrict__ edst,
                          const float* __restrict__ mu, float* __restrict__ nb,
                          float* __restrict__ deg, int E) {
  int gwave = (blockIdx.x * blockDim.x + threadIdx.x) >> 5;
  int lane = threadIdx.x & 31;
  int nwaves = (gridDim.x * blockDim.x) >> 5;
  for (int e = gwave; e < E; e += nwaves) {
    int src = esrc[e], dst = edst[e];
    v4f v = *(const v4f*)(mu + (size_t)dst * NP + lane * 4);
    float* p = nb + (size_t)src * NP + lane * 4;
    __hip_atomic_fetch_add(p + 0, v.x, __ATOMIC_RELAXED, __HIP_MEMORY_SCOPE_AGENT);
    __hip_atomic_fetch_add(p + 1, v.y, __ATOMIC_RELAXED, __HIP_MEMORY_SCOPE_AGENT);
    __hip_atomic_fetch_add(p + 2, v.z, __ATOMIC_RELAXED, __HIP_MEMORY_SCOPE_AGENT);
    __hip_atomic_fetch_add(p + 3, v.w, __ATOMIC_RELAXED, __HIP_MEMORY_SCOPE_AGENT);
    if (lane == 0)
      __hip_atomic_fetch_add(deg + src, 1.0f, __ATOMIC_RELAXED, __HIP_MEMORY_SCOPE_AGENT);
  }
}

// mu' = relu([mu ; deg*mu - nb]) @ h_theta^T  (K = 256, bf16 A built in-register)
// Also accumulates column sums of mu' into sum_mp via LDS + one atomic/col/block.
__global__ void k_muprime(const float* __restrict__ mu, const float* __restrict__ nb,
                          const float* __restrict__ deg, const __bf16* __restrict__ hth,
                          __bf16* __restrict__ mup, float* __restrict__ sum_mp, int ntiles) {
  __shared__ float lsum[NP];
  if (threadIdx.x < NP) lsum[threadIdx.x] = 0.f;
  __syncthreads();

  int tile = blockIdx.x * (blockDim.x >> 5) + (threadIdx.x >> 5);
  int lane = threadIdx.x & 31;
  int half = lane >> 4, l15 = lane & 15;
  bool valid = tile < ntiles;
  float colsum[8] = {};

  if (valid) {
    int m0 = tile * 16;
    int rowA = m0 + l15;
    float dg = deg[rowA];
    const float* mrow = mu + (size_t)rowA * NP;
    const float* nrow = nb + (size_t)rowA * NP;
    v8f acc[8] = {};
    #pragma unroll
    for (int kc = 0; kc < 8; ++kc) {
      int k0 = kc * 32;
      v16bf a;
      if (k0 < NP) {                       // relu(mu) half
        int ko = k0 + half * 16;
        put4(a, 0,  relu4(*(const v4f*)(mrow + ko)));
        put4(a, 4,  relu4(*(const v4f*)(mrow + ko + 4)));
        put4(a, 8,  relu4(*(const v4f*)(mrow + ko + 8)));
        put4(a, 12, relu4(*(const v4f*)(mrow + ko + 12)));
      } else {                             // relu(deg*mu - nb) half
        int ko = (k0 - NP) + half * 16;
        v4f a0 = *(const v4f*)(mrow + ko),      b0 = *(const v4f*)(nrow + ko);
        v4f a1 = *(const v4f*)(mrow + ko + 4),  b1 = *(const v4f*)(nrow + ko + 4);
        v4f a2 = *(const v4f*)(mrow + ko + 8), b2 = *(const v4f*)(nrow + ko + 8);
        v4f a3 = *(const v4f*)(mrow + ko + 12), b3 = *(const v4f*)(nrow + ko + 12);
        put4(a, 0,  relu4(a0 * dg - b0));
        put4(a, 4,  relu4(a1 * dg - b1));
        put4(a, 8,  relu4(a2 * dg - b2));
        put4(a, 12, relu4(a3 * dg - b3));
      }
      #pragma unroll
      for (int t = 0; t < 8; ++t) {
        v16bf b = *(const v16bf*)(hth + (size_t)(t * 16 + l15) * N2P + k0 + half * 16);
        acc[t] = __builtin_amdgcn_wmma_f32_16x16x32_bf16(
            false, a, false, b, (short)0, acc[t], false, false);
      }
    }
    int rbase = m0 + half * 8;
    #pragma unroll
    for (int t = 0; t < 8; ++t) {
      int c = t * 16 + l15;
      #pragma unroll
      for (int r = 0; r < 8; ++r) {
        float val = acc[t][r];
        mup[(size_t)(rbase + r) * NP + c] = (__bf16)val;
        colsum[t] += val;
      }
    }
  }
  if (valid) {
    #pragma unroll
    for (int t = 0; t < 8; ++t) atomicAdd(&lsum[t * 16 + l15], colsum[t]);
  }
  __syncthreads();
  if (threadIdx.x < NP)
    __hip_atomic_fetch_add(&sum_mp[threadIdx.x], lsum[threadIdx.x],
                           __ATOMIC_RELAXED, __HIP_MEMORY_SCOPE_AGENT);
}

// up = theta4 @ sum_mp; c0 = theta3[0:P] . relu(up)   (one block of 128 threads)
__global__ void k_up(const float* __restrict__ th4, const float* __restrict__ th3,
                     const float* __restrict__ sum_mp, float* __restrict__ c0) {
  __shared__ float red[NP];
  int p = threadIdx.x;
  float acc = 0.f;
  for (int k = 0; k < NP; ++k) acc += th4[p * NP + k] * sum_mp[k];
  red[p] = relu_f(acc) * th3[p];
  __syncthreads();
  for (int s = 64; s >= 1; s >>= 1) {
    if (p < s) red[p] += red[p + s];
    __syncthreads();
  }
  if (p == 0) c0[0] = red[0];
}

// down = mu' @ theta5^T; out = c0 + theta3[P:2P] . relu(down_v)
__global__ void k_out(const __bf16* __restrict__ mup, const __bf16* __restrict__ t5,
                      const float* __restrict__ th3, const float* __restrict__ c0p,
                      float* __restrict__ out, int ntiles) {
  int tile = blockIdx.x * (blockDim.x >> 5) + (threadIdx.x >> 5);
  if (tile >= ntiles) return;
  int lane = threadIdx.x & 31;
  int half = lane >> 4, l15 = lane & 15;
  int m0 = tile * 16;
  int rowA = m0 + l15;
  v8f acc[8] = {};
  #pragma unroll
  for (int kc = 0; kc < 4; ++kc) {
    int k0 = kc * 32;
    const v8bf* pa = (const v8bf*)(mup + (size_t)rowA * NP + k0 + half * 16);
    v16bf a = __builtin_shufflevector(pa[0], pa[1],
        0,1,2,3,4,5,6,7,8,9,10,11,12,13,14,15);
    #pragma unroll
    for (int t = 0; t < 8; ++t) {
      v16bf b = *(const v16bf*)(t5 + (size_t)(t * 16 + l15) * NP + k0 + half * 16);
      acc[t] = __builtin_amdgcn_wmma_f32_16x16x32_bf16(
          false, a, false, b, (short)0, acc[t], false, false);
    }
  }
  float rowacc[8] = {};
  #pragma unroll
  for (int t = 0; t < 8; ++t) {
    float t3 = th3[NP + t * 16 + l15];
    #pragma unroll
    for (int r = 0; r < 8; ++r) rowacc[r] += relu_f(acc[t][r]) * t3;
  }
  #pragma unroll
  for (int r = 0; r < 8; ++r) {
    #pragma unroll
    for (int m = 8; m >= 1; m >>= 1) rowacc[r] += __shfl_xor(rowacc[r], m, 16);
  }
  if (l15 == 0) {
    float c0 = c0p[0];
    int rbase = m0 + half * 8;
    #pragma unroll
    for (int r = 0; r < 8; ++r) out[rbase + r] = c0 + rowacc[r];
  }
}

// ---------------------------------------------------------------- launch
extern "C" void kernel_launch(void* const* d_in, const int* in_sizes, int n_in,
                              void* d_out, int out_size, void* d_ws, size_t ws_size,
                              hipStream_t stream) {
  const float* s_v = (const float*)d_in[0];
  const float* t_v = (const float*)d_in[1];
  const float* x   = (const float*)d_in[2];
  const int*   esr = (const int*)  d_in[3];
  const int*   eds = (const int*)  d_in[4];
  const float* th1 = (const float*)d_in[5];
  const float* th2 = (const float*)d_in[6];
  const float* th3 = (const float*)d_in[7];
  const float* th4 = (const float*)d_in[8];
  const float* th5 = (const float*)d_in[9];
  const float* hth = (const float*)d_in[10];
  float* out = (float*)d_out;

  const int N = in_sizes[0];
  const int E = in_sizes[3];
  const int ntiles = (N + 15) / 16;   // N=100000 -> 6250 (exact)

  // workspace layout (~128 MB)
  char* ws = (char*)d_ws;
  size_t off = 0;
  auto carve = [&](size_t bytes) {
    void* p = ws + off;
    off += (bytes + 255) & ~(size_t)255;
    return p;
  };
  float*  mu     = (float*) carve((size_t)N * NP * 4);
  float*  nb     = (float*) carve((size_t)N * NP * 4);
  __bf16* xn     = (__bf16*)carve((size_t)N * NK * 2);  // reused as mu' (bf16) later
  __bf16* mup    = xn;                                   // overlay: xn dead after k_mu
  float*  deg    = (float*) carve((size_t)N * 4);
  float*  sum_mp = (float*) carve(NP * 4);
  float*  c0     = (float*) carve(4);
  __bf16* t2bf   = (__bf16*)carve((size_t)NP * NK * 2);
  __bf16* t5bf   = (__bf16*)carve((size_t)NP * NP * 2);
  __bf16* hthbf  = (__bf16*)carve((size_t)NP * N2P * 2);
  (void)ws_size; (void)n_in; (void)out_size;

  // zero accumulator buffers (graph-capturable async memsets)
  hipMemsetAsync(nb,     0, (size_t)N * NP * 4, stream);
  hipMemsetAsync(deg,    0, (size_t)N * 4,      stream);
  hipMemsetAsync(sum_mp, 0, NP * 4,             stream);

  // weight conversions
  k_cvt_bf16<<<(NP * NK + 255) / 256, 256, 0, stream>>>(th2, t2bf, NP * NK);
  k_cvt_bf16<<<(NP * NP + 255) / 256, 256, 0, stream>>>(th5, t5bf, NP * NP);
  k_cvt_bf16<<<(NP * N2P + 255) / 256, 256, 0, stream>>>(hth, hthbf, NP * N2P);

  // stage 1: normalize rows of x
  k_norm<<<(N * 32 + 255) / 256, 256, 0, stream>>>(x, xn, N);

  // stage 2: mu (WMMA)
  int gemm_blocks = (ntiles + 7) / 8;   // 8 waves / block
  k_mu<<<gemm_blocks, 256, 0, stream>>>(s_v, t_v, xn, t2bf, th1, mu, ntiles);

  // stage 3: edge scatter (dominant, ~3.2 GB of HBM traffic)
  k_scatter<<<2048, 256, 0, stream>>>(esr, eds, mu, nb, deg, E);

  // stage 4: mu' (WMMA, K=256) + column-sum reduction
  k_muprime<<<gemm_blocks, 256, 0, stream>>>(mu, nb, deg, hthbf, mup, sum_mp, ntiles);

  // stage 5: shared scalar c0 = theta3_top . relu(theta4 @ sum_mu')
  k_up<<<1, NP, 0, stream>>>(th4, th3, sum_mp, c0);

  // stage 6: down GEMM (WMMA) + fused final dot -> out
  k_out<<<gemm_blocks, 256, 0, stream>>>(mup, t5bf, th3, c0, out, ntiles);
}
